// CausalSelfAttention_2156073583292
// MI455X (gfx1250) — compile-verified, matrix operand order checked by reading, not run on hardware
//
#include <hip/hip_runtime.h>
#include <hip/hip_bf16.h>

typedef __attribute__((ext_vector_type(16))) __bf16 v16bf;
typedef __attribute__((ext_vector_type(2)))  __bf16 v2bf;
typedef __attribute__((ext_vector_type(8)))  float  v8f;
typedef __attribute__((ext_vector_type(2)))  float  v2f;

#define N_B 256
#define N_T 256
#define N_C 192
#define N_H 6
#define N_D 32
#define MROWS (N_B * N_T)

// Packed f32x2 -> bf16x2 via vector convert (selects v_cvt_pk_bf16_f32)
__device__ __forceinline__ unsigned int packcvt(float lo, float hi) {
    v2f f = {lo, hi};
    v2bf b = __builtin_convertvector(f, v2bf);
    union { v2bf v; unsigned int u; } c;
    c.v = b;
    return c.u;
}
__device__ __forceinline__ unsigned short bf16bits(float f) {
    __bf16 h = (__bf16)f;
    union { __bf16 b; unsigned short s; } c;
    c.b = h;
    return c.s;
}

// A/B operand: per lane two contiguous 16B runs (elements [8h,8h+8) and [16+8h,16+8h+8))
union ABv { v16bf v; uint4 q[2]; unsigned int u[8]; };

// ---------------------------------------------------------------------------
// Software-pipelined 16x64 GEMM core: 6 K-chunks x 4 N-tiles, prefetch next
// operands before each WMMA so loads overlap matrix ops.
// ---------------------------------------------------------------------------
__device__ __forceinline__ void gemm_16x64_core(
        const uint4* __restrict__ pA, const uint4* const* __restrict__ pB,
        v8f acc[4]) {
    ABv a_cur, b_cur, a_nxt, b_nxt;
    a_cur.q[0] = pA[0];
    a_cur.q[1] = pA[2];
    b_cur.q[0] = pB[0][0];
    b_cur.q[1] = pB[0][2];
    #pragma unroll
    for (int kk = 0; kk < 6; ++kk) {
        const int qi = kk * 4;                   // uint4 index of this K-chunk
        #pragma unroll
        for (int t = 0; t < 4; ++t) {
            if (t < 3) {                         // prefetch next N-tile
                b_nxt.q[0] = pB[t + 1][qi];
                b_nxt.q[1] = pB[t + 1][qi + 2];
            } else if (kk < 5) {                 // prefetch next K-chunk
                b_nxt.q[0] = pB[0][qi + 4];
                b_nxt.q[1] = pB[0][qi + 6];
                a_nxt.q[0] = pA[qi + 4];
                a_nxt.q[1] = pA[qi + 6];
            }
            acc[t] = __builtin_amdgcn_wmma_f32_16x16x32_bf16(
                false, a_cur.v, false, b_cur.v, (short)0, acc[t], false, false);
            b_cur = b_nxt;
            if (t == 3) a_cur = a_nxt;
        }
    }
}

// ---------------------------------------------------------------------------
// Kernel 0: bulk f32 -> bf16 conversion (float4 in, uint2 out, coalesced)
// ---------------------------------------------------------------------------
__global__ __launch_bounds__(256) void cvt_bf16(
        const float4* __restrict__ src, uint2* __restrict__ dst, int n4) {
    const int i = blockIdx.x * 256 + threadIdx.x;
    if (i < n4) {
        const float4 f = src[i];
        uint2 o;
        o.x = packcvt(f.x, f.y);
        o.y = packcvt(f.z, f.w);
        dst[i] = o;
    }
}

// ---------------------------------------------------------------------------
// Kernel 1: QKV projection.  QKV[r, c] = X[r, :] . Wqkv[c, :]  (both bf16)
// Scatters bf16 results into Q[b,h,t,d], K[b,h,t,d], Vt[b,h,d,t].
// ---------------------------------------------------------------------------
__global__ __launch_bounds__(128) void qkv_gemm(
        const unsigned short* __restrict__ Xb, const unsigned short* __restrict__ Wb,
        unsigned short* __restrict__ Q, unsigned short* __restrict__ K,
        unsigned short* __restrict__ Vt) {
    const int lane = threadIdx.x & 31;
    const int wave = threadIdx.x >> 5;
    const int l15  = lane & 15;
    const int half = lane >> 4;
    const int rowBase = blockIdx.x * 64 + wave * 16;
    const int colBase = blockIdx.y * 64;

    v8f acc[4] = {{}, {}, {}, {}};
    const int r = rowBase + l15;

    const uint4* pA = (const uint4*)(Xb + (size_t)r * N_C + 8 * half);
    const uint4* pB[4] = {
        (const uint4*)(Wb + (size_t)(colBase +  0 + l15) * N_C + 8 * half),
        (const uint4*)(Wb + (size_t)(colBase + 16 + l15) * N_C + 8 * half),
        (const uint4*)(Wb + (size_t)(colBase + 32 + l15) * N_C + 8 * half),
        (const uint4*)(Wb + (size_t)(colBase + 48 + l15) * N_C + 8 * half)};

    gemm_16x64_core(pA, pB, acc);

    const int sel    = colBase / N_C;        // uniform per block
    const int ccBase = colBase % N_C;        // multiple of 64
    const int b  = rowBase >> 8;             // uniform per block
    #pragma unroll
    for (int t = 0; t < 4; ++t) {
        const int h = (ccBase + t * 16) >> 5;            // uniform per t
        const int d = ((ccBase + t * 16) & 31) + l15;
        #pragma unroll
        for (int j = 0; j < 8; ++j) {
            const int tt = (rowBase + j + 8 * half) & 255;
            const unsigned short bits = bf16bits(acc[t][j]);
            if (sel == 0)
                Q[(((size_t)(b * N_H + h)) * N_T + tt) * N_D + d] = bits;
            else if (sel == 1)
                K[(((size_t)(b * N_H + h)) * N_T + tt) * N_D + d] = bits;
            else
                Vt[(((size_t)(b * N_H + h)) * N_D + d) * N_T + tt] = bits;
        }
    }
}

// ---------------------------------------------------------------------------
// Kernel 2: flash-style causal attention.  One wave = 16 query rows of one
// (b,h).  D=32 == one bf16-WMMA K step.  32-key blocks: 2 WMMA (QK^T) +
// online softmax + 2 WMMA (PV).  V loads are hoisted above the softmax so
// their latency hides under the VALU work.
// ---------------------------------------------------------------------------
__global__ __launch_bounds__(128) void attn(
        const unsigned short* __restrict__ Q, const unsigned short* __restrict__ K,
        const unsigned short* __restrict__ Vt, unsigned short* __restrict__ Y) {
    __shared__ float plds[4][16 * 36];          // per-wave P staging (stride 36)
    const int lane = threadIdx.x & 31;
    const int wave = threadIdx.x >> 5;
    const int l15  = lane & 15;
    const int half = lane >> 4;
    const int bh = blockIdx.x;                  // b*H + h
    const int qs = blockIdx.y * 64 + wave * 16; // this wave's first query row

    // Q A-operand (16 rows x K=32) -- loaded once, reused every key block
    ABv aQ;
    {
        const uint4* pQ = (const uint4*)(Q + ((size_t)bh * N_T + qs + l15) * N_D + 8 * half);
        aQ.q[0] = pQ[0];
        aQ.q[1] = pQ[2];
    }

    // per-lane operand base pointers with explicit induction over key blocks
    const uint4* pK = (const uint4*)(K  + ((size_t)bh * N_T + l15) * N_D + 8 * half);
    const uint4* pV = (const uint4*)(Vt + ((size_t)bh * N_D + l15) * N_T + 8 * half);

    v8f o0 = {}, o1 = {};
    float m_run[8], l_run[8];
    #pragma unroll
    for (int j = 0; j < 8; ++j) { m_run[j] = -1e30f; l_run[j] = 0.f; }
    const float scale = 0.17677669529663687f;   // 1/sqrt(32)
    const int nkb = (qs + 47) >> 5;             // causal trip count

    for (int kb = 0; kb < nkb; ++kb) {
        // ---- scores S = (Q K^T) * scale, two 16x16 tiles ----
        v8f s0 = {}, s1 = {};
        {
            ABv bK;
            bK.q[0] = pK[0];          // keys kb*32 + l15
            bK.q[1] = pK[2];
            s0 = __builtin_amdgcn_wmma_f32_16x16x32_bf16(
                false, aQ.v, false, bK.v, (short)0, s0, false, false);
            bK.q[0] = pK[64];         // +16 rows (16*32 elems / 8)
            bK.q[1] = pK[66];
            s1 = __builtin_amdgcn_wmma_f32_16x16x32_bf16(
                false, aQ.v, false, bK.v, (short)0, s1, false, false);
        }
        // ---- issue V loads now; consumed only after the softmax ----
        ABv bV0, bV1;
        bV0.q[0] = pV[0];             // dims l15,    keys kb*32..
        bV0.q[1] = pV[2];
        bV1.q[0] = pV[512];           // dims 16+l15 (16*256 elems / 8)
        bV1.q[1] = pV[514];

        #pragma unroll
        for (int j = 0; j < 8; ++j) { s0[j] *= scale; s1[j] *= scale; }
        if (kb * 32 + 31 > qs) {                // partial (diagonal) block
            #pragma unroll
            for (int j = 0; j < 8; ++j) {
                const int q = qs + j + 8 * half;
                const int k0 = kb * 32 + l15;
                if (k0 > q)      s0[j] = -1e30f;
                if (k0 + 16 > q) s1[j] = -1e30f;
            }
        }
        // ---- online softmax (rows are fixed VGPR slots; reduce over lanes) ----
        float p0[8], p1[8];
        #pragma unroll
        for (int j = 0; j < 8; ++j) {
            float vmax = fmaxf(s0[j], s1[j]);
            vmax = fmaxf(vmax, __shfl_xor(vmax, 8));
            vmax = fmaxf(vmax, __shfl_xor(vmax, 4));
            vmax = fmaxf(vmax, __shfl_xor(vmax, 2));
            vmax = fmaxf(vmax, __shfl_xor(vmax, 1));
            const float mnew = fmaxf(m_run[j], vmax);
            const float corr = __expf(m_run[j] - mnew);
            m_run[j] = mnew;
            p0[j] = __expf(s0[j] - mnew);
            p1[j] = __expf(s1[j] - mnew);
            float rs = p0[j] + p1[j];
            rs += __shfl_xor(rs, 8);
            rs += __shfl_xor(rs, 4);
            rs += __shfl_xor(rs, 2);
            rs += __shfl_xor(rs, 1);
            l_run[j] = l_run[j] * corr + rs;
            o0[j] *= corr;
            o1[j] *= corr;
        }
        // ---- P: C/D layout -> LDS -> A-operand layout (wave-local) ----
        float* L = plds[wave];
        #pragma unroll
        for (int j = 0; j < 8; ++j) {
            const int row = j + 8 * half;
            L[row * 36 + l15]      = p0[j];
            L[row * 36 + 16 + l15] = p1[j];
        }
        asm volatile("s_wait_dscnt 0x0" ::: "memory");
        __builtin_amdgcn_wave_barrier();
        ABv aP;
        {
            const float* Lr = L + l15 * 36 + 8 * half;
            const float4 r0 = *(const float4*)(Lr);          // K 8h..8h+3
            const float4 r1 = *(const float4*)(Lr + 4);      // K 8h+4..8h+7
            const float4 r2 = *(const float4*)(Lr + 16);     // K 16+8h..
            const float4 r3 = *(const float4*)(Lr + 20);
            aP.u[0] = packcvt(r0.x, r0.y);
            aP.u[1] = packcvt(r0.z, r0.w);
            aP.u[2] = packcvt(r1.x, r1.y);
            aP.u[3] = packcvt(r1.z, r1.w);
            aP.u[4] = packcvt(r2.x, r2.y);
            aP.u[5] = packcvt(r2.z, r2.w);
            aP.u[6] = packcvt(r3.x, r3.y);
            aP.u[7] = packcvt(r3.z, r3.w);
        }
        asm volatile("s_wait_dscnt 0x0" ::: "memory");
        __builtin_amdgcn_wave_barrier();
        // ---- O += P @ V ----
        o0 = __builtin_amdgcn_wmma_f32_16x16x32_bf16(
            false, aP.v, false, bV0.v, (short)0, o0, false, false);
        o1 = __builtin_amdgcn_wmma_f32_16x16x32_bf16(
            false, aP.v, false, bV1.v, (short)0, o1, false, false);

        pK += 128;                    // +32 key rows (32*32 elems / 8)
        pV += 4;                      // +32 key elems  (32 / 8)
    }

    // ---- normalize + store Y[b,t,h*32+d] as bf16 ----
    const int b = bh / N_H, h = bh % N_H;
    #pragma unroll
    for (int j = 0; j < 8; ++j) {
        const int q = qs + j + 8 * half;
        const float inv = 1.0f / l_run[j];
        const size_t base = ((size_t)(b * N_T + q)) * N_C + h * N_D;
        Y[base + l15]      = bf16bits(o0[j] * inv);
        Y[base + 16 + l15] = bf16bits(o1[j] * inv);
    }
}

// ---------------------------------------------------------------------------
// Kernel 3: output projection.  Out[r, c] = Y[r, :] . Wout[c, :]  (bf16 x bf16)
// ---------------------------------------------------------------------------
__global__ __launch_bounds__(128) void out_gemm(
        const unsigned short* __restrict__ Yb, const unsigned short* __restrict__ Wb,
        float* __restrict__ Out) {
    const int lane = threadIdx.x & 31;
    const int wave = threadIdx.x >> 5;
    const int l15  = lane & 15;
    const int half = lane >> 4;
    const int rowBase = blockIdx.x * 64 + wave * 16;
    const int colBase = blockIdx.y * 64;

    v8f acc[4] = {{}, {}, {}, {}};
    const int r = rowBase + l15;

    const uint4* pA = (const uint4*)(Yb + (size_t)r * N_C + 8 * half);
    const uint4* pB[4] = {
        (const uint4*)(Wb + (size_t)(colBase +  0 + l15) * N_C + 8 * half),
        (const uint4*)(Wb + (size_t)(colBase + 16 + l15) * N_C + 8 * half),
        (const uint4*)(Wb + (size_t)(colBase + 32 + l15) * N_C + 8 * half),
        (const uint4*)(Wb + (size_t)(colBase + 48 + l15) * N_C + 8 * half)};

    gemm_16x64_core(pA, pB, acc);

    #pragma unroll
    for (int t = 0; t < 4; ++t) {
        #pragma unroll
        for (int j = 0; j < 8; ++j) {
            const int r2 = rowBase + j + 8 * half;
            const int c  = colBase + t * 16 + l15;
            Out[(size_t)r2 * N_C + c] = acc[t][j];
        }
    }
}

// ---------------------------------------------------------------------------
extern "C" void kernel_launch(void* const* d_in, const int* in_sizes, int n_in,
                              void* d_out, int out_size, void* d_ws, size_t ws_size,
                              hipStream_t stream) {
    const float* x     = (const float*)d_in[0];
    const float* w_qkv = (const float*)d_in[1];
    const float* w_out = (const float*)d_in[2];
    float* out = (float*)d_out;

    const size_t nQKV = (size_t)N_B * N_H * N_T * N_D;   // 12,582,912 elems
    const size_t nX   = (size_t)MROWS * N_C;             // 12,582,912 elems
    const size_t nWq  = (size_t)3 * N_C * N_C;           // 110,592
    const size_t nWo  = (size_t)N_C * N_C;               // 36,864

    unsigned short* Q   = (unsigned short*)d_ws;
    unsigned short* K   = Q   + nQKV;
    unsigned short* Vt  = K   + nQKV;
    unsigned short* Y   = Vt  + nQKV;
    unsigned short* Xb  = Y   + nQKV;
    unsigned short* Wqb = Xb  + nX;
    unsigned short* Wob = Wqb + nWq;

    // ---- stage inputs to bf16 once ----
    cvt_bf16<<<dim3((unsigned)(nX / 4 / 256)), 256, 0, stream>>>(
        (const float4*)x, (uint2*)Xb, (int)(nX / 4));
    cvt_bf16<<<dim3((unsigned)(nWq / 4 / 256)), 256, 0, stream>>>(
        (const float4*)w_qkv, (uint2*)Wqb, (int)(nWq / 4));
    cvt_bf16<<<dim3((unsigned)((nWo / 4 + 255) / 256)), 256, 0, stream>>>(
        (const float4*)w_out, (uint2*)Wob, (int)(nWo / 4));

    dim3 g1(MROWS / 64, (3 * N_C) / 64);   // (1024, 9)
    qkv_gemm<<<g1, 128, 0, stream>>>(Xb, Wqb, Q, K, Vt);

    dim3 g2(N_B * N_H, N_T / 64);          // (1536, 4)
    attn<<<g2, 128, 0, stream>>>(Q, K, Vt, Y);

    dim3 g3(MROWS / 64, N_C / 64);         // (1024, 3)
    out_gemm<<<g3, 128, 0, stream>>>(Y, Wob, out);
}
